// ScaledDotProductAttention_63153199120542
// MI455X (gfx1250) — matrix operand order repaired and checked by reading, not compiled, not run on hardware
//
#include <hip/hip_runtime.h>
#include <hip/hip_bf16.h>

typedef __attribute__((ext_vector_type(16))) __bf16 v16bf;
typedef __attribute__((ext_vector_type(8)))  __bf16 v8bf;
typedef __attribute__((ext_vector_type(8)))  float  v8f;
typedef __attribute__((ext_vector_type(4)))  unsigned int v4u;
typedef __attribute__((ext_vector_type(8)))  int    v8i;
typedef __attribute__((ext_vector_type(4)))  int    v4i;

#define S_LEN   2048
#define DK_     64
#define NBH     48          // B*H
#define QBLK    128         // Q rows per workgroup (8 waves x 16 rows)
#define KSTRIDE 72          // K LDS row stride, bf16 elems (64 + 4-DWORD TDM pad)
#define VSTRIDE 40          // V^T LDS row stride, bf16 elems (32 + 4-DWORD TDM pad)
#define PSTRIDE 40          // P staging row stride, bf16 elems
#define NKT     64          // number of 32-key tiles

#if defined(__gfx1250__) && __has_builtin(__builtin_amdgcn_tensor_load_to_lds)
#define USE_TDM 1
#else
#define USE_TDM 0
#endif

#if defined(__gfx1250__) && __has_builtin(__builtin_amdgcn_mov_dpp8) && __has_builtin(__builtin_amdgcn_mov_dpp)
#define USE_DPP 1
#else
#define USE_DPP 0
#endif

// ---------------- fast math helpers ----------------
__device__ __forceinline__ float fast_exp2(float x) {
#if defined(__gfx1250__) && __has_builtin(__builtin_amdgcn_exp2f)
    return __builtin_amdgcn_exp2f(x);       // native v_exp_f32
#else
    return exp2f(x);
#endif
}
__device__ __forceinline__ float fast_rcp(float x) {
#if defined(__gfx1250__) && __has_builtin(__builtin_amdgcn_rcpf)
    return __builtin_amdgcn_rcpf(x);        // native v_rcp_f32
#else
    return 1.0f / x;
#endif
}

// ---------------- cross-lane reductions over each 16-lane half ----------------
#if USE_DPP
constexpr int dpp8_sel(int a,int b,int c,int d,int e,int f,int g,int h){
    return a|(b<<3)|(c<<6)|(d<<9)|(e<<12)|(f<<15)|(g<<18)|(h<<21);
}
constexpr int DPP_XOR1 = dpp8_sel(1,0,3,2,5,4,7,6);
constexpr int DPP_XOR2 = dpp8_sel(2,3,0,1,6,7,4,5);
constexpr int DPP_XOR4 = dpp8_sel(4,5,6,7,0,1,2,3);
template<int SEL>
__device__ __forceinline__ float dpp8f(float x) {
    return __int_as_float(__builtin_amdgcn_mov_dpp8(__float_as_int(x), SEL));
}
__device__ __forceinline__ float dppmirf(float x) {   // DPP16 ROW_MIRROR (0x140)
    return __int_as_float(__builtin_amdgcn_mov_dpp(__float_as_int(x), 0x140, 0xf, 0xf, true));
}
#endif

__device__ __forceinline__ float rowmax16(float x) {
#if USE_DPP
    x = fmaxf(x, dpp8f<DPP_XOR1>(x));
    x = fmaxf(x, dpp8f<DPP_XOR2>(x));
    x = fmaxf(x, dpp8f<DPP_XOR4>(x));
    x = fmaxf(x, dppmirf(x));          // 8-groups uniform -> mirror merges halves of 16
#else
    x = fmaxf(x, __shfl_xor(x, 1, 32));
    x = fmaxf(x, __shfl_xor(x, 2, 32));
    x = fmaxf(x, __shfl_xor(x, 4, 32));
    x = fmaxf(x, __shfl_xor(x, 8, 32));
#endif
    return x;
}
__device__ __forceinline__ float rowsum16(float x) {
#if USE_DPP
    x += dpp8f<DPP_XOR1>(x);
    x += dpp8f<DPP_XOR2>(x);
    x += dpp8f<DPP_XOR4>(x);
    x += dppmirf(x);
#else
    x += __shfl_xor(x, 1, 32);
    x += __shfl_xor(x, 2, 32);
    x += __shfl_xor(x, 4, 32);
    x += __shfl_xor(x, 8, 32);
#endif
    return x;
}

__device__ __forceinline__ v8f wmma_bf16(v16bf a, v16bf b, v8f c) {
    return __builtin_amdgcn_wmma_f32_16x16x32_bf16(false, a, false, b, (short)0, c, false, false);
}

#if USE_TDM
// 2D TDM load: tile_d1 rows of tile_d0 bf16 elements, global row stride stride0 elems,
// LDS padding (pad_amount code) DWORDs after every (pad_interval code) DWORDs.
__device__ __forceinline__ void tdm_load_2d(unsigned int lds_byte, const void* gptr,
                                            unsigned int tile_d0, unsigned int tile_d1,
                                            unsigned int stride0,
                                            unsigned int pad_interval, unsigned int pad_amount)
{
    const unsigned long long ga = (unsigned long long)(uintptr_t)gptr;
    v4u g0;
    g0.x = 1u;                                                    // count=1, user D#
    g0.y = lds_byte;                                              // lds_addr
    g0.z = (unsigned int)ga;                                      // global_addr[31:0]
    g0.w = (unsigned int)((ga >> 32) & 0x01FFFFFFu) | (2u << 30); // global_addr[56:32] | type=2
    v8i g1;
    g1[0] = (int)((1u << 16) | (1u << 20) | (pad_interval << 22) | (pad_amount << 25)); // data_size=2B, pad_en
    g1[1] = (int)(tile_d0 << 16);            // tensor_dim0[15:0]  (== tile_d0)
    g1[2] = (int)(tile_d1 << 16);            // tensor_dim0 hi=0 | tensor_dim1[15:0]
    g1[3] = (int)(tile_d0 << 16);            // tensor_dim1 hi=0 | tile_dim0
    g1[4] = (int)(tile_d1 & 0xffffu);        // tile_dim1 | tile_dim2=0
    g1[5] = (int)stride0;                    // tensor_dim0_stride[31:0]
    g1[6] = 0;
    g1[7] = 0;
    v4i z4 = {0, 0, 0, 0};
#if __clang_major__ >= 23
    v8i z8 = {0,0,0,0,0,0,0,0};
    __builtin_amdgcn_tensor_load_to_lds(g0, g1, z4, z4, z8, 0);
#else
    __builtin_amdgcn_tensor_load_to_lds(g0, g1, z4, z4, 0);
#endif
}
__device__ __forceinline__ void wait_tensor0() {
#if __has_builtin(__builtin_amdgcn_s_wait_tensorcnt)
    __builtin_amdgcn_s_wait_tensorcnt(0);
#else
    asm volatile("s_wait_tensorcnt 0x0" ::: "memory");
#endif
}
#endif // USE_TDM

// ================= prep kernel: K -> bf16, V -> bf16 transposed (d-major) =================
__launch_bounds__(256)
__global__ void sdpa_prep_kernel(const float* __restrict__ K, const float* __restrict__ V,
                                 __bf16* __restrict__ Kbf, __bf16* __restrict__ Vt)
{
    __shared__ float tile[64][65];
    const int bh  = blockIdx.x >> 5;     // 0..47
    const int kt  = blockIdx.x & 31;     // 64-key tile
    const int tid = threadIdx.x;
    const size_t base = (size_t)bh * S_LEN * DK_ + (size_t)kt * 64 * DK_;

    #pragma unroll
    for (int j = 0; j < 16; ++j) {       // K convert, coalesced
        const int e = tid + j * 256;
        Kbf[base + e] = (__bf16)K[base + e];
    }
    #pragma unroll
    for (int j = 0; j < 16; ++j) {       // V 64x64 tile into LDS
        const int e = tid + j * 256;
        tile[e >> 6][e & 63] = V[base + e];
    }
    __syncthreads();
    __bf16* vt = Vt + (size_t)bh * DK_ * S_LEN;       // [64 d][2048 keys]
    #pragma unroll
    for (int j = 0; j < 16; ++j) {       // transposed write, coalesced over keys
        const int e = tid + j * 256;
        const int d = e >> 6, key = e & 63;
        vt[(size_t)d * S_LEN + kt * 64 + key] = (__bf16)tile[key][d];
    }
}

// ================= fused attention kernel =================
__launch_bounds__(256)
__global__ void sdpa_fused_kernel(const float* __restrict__ Q,
                                  const __bf16* __restrict__ Kbf,
                                  const __bf16* __restrict__ Vt,
                                  float* __restrict__ ctx,
                                  float* __restrict__ attn)
{
    __shared__ __attribute__((aligned(32))) __bf16 Klds[2][32 * KSTRIDE]; // 32 keys x 64 d (padded)
    __shared__ __attribute__((aligned(32))) __bf16 Vlds[2][64 * VSTRIDE]; // 64 d x 32 keys (padded)
    __shared__ __attribute__((aligned(32))) __bf16 Plds[8 * 16 * PSTRIDE];

    const int bh   = blockIdx.x >> 4;
    const int qb   = blockIdx.x & 15;
    const int tid  = threadIdx.x;
    const int lane = tid & 31;
    const int wave = tid >> 5;
    const int half = lane >> 4;
    const int l16  = lane & 15;

    // base-2 softmax: fold log2(e) into the score scale; softmax is invariant
    const float scale = 0.125f * 1.44269504088896340736f;   // (1/sqrt(64)) * log2(e)

    const float*  Qp  = Q   + (size_t)bh * S_LEN * DK_;
    const __bf16* Kbh = Kbf + (size_t)bh * S_LEN * DK_;
    const __bf16* Vth = Vt  + (size_t)bh * DK_ * S_LEN;
    float*        Cp  = ctx  + (size_t)bh * S_LEN * DK_;
    float*        Ap  = attn + (size_t)bh * S_LEN * S_LEN;

    const int qrow0 = qb * QBLK + wave * 16;
    const int myrow = qrow0 + l16;

    // ---- Q rows into WMMA A-layout registers, pre-scaled (vectorized 32B runs) ----
    v16bf qa0, qa1;
    {
        const float* qr = Qp + (size_t)myrow * DK_;
        const v8f qlo0 = *(const v8f*)(qr +      half * 8);   // chunk0, v=0..3 pairs
        const v8f qhi0 = *(const v8f*)(qr + 16 + half * 8);   // chunk0, v=4..7 pairs
        const v8f qlo1 = *(const v8f*)(qr + 32 + half * 8);   // chunk1, v=0..3 pairs
        const v8f qhi1 = *(const v8f*)(qr + 48 + half * 8);   // chunk1, v=4..7 pairs
        #pragma unroll
        for (int j = 0; j < 8; ++j) {
            qa0[j]     = (__bf16)(qlo0[j] * scale);
            qa0[8 + j] = (__bf16)(qhi0[j] * scale);
            qa1[j]     = (__bf16)(qlo1[j] * scale);
            qa1[8 + j] = (__bf16)(qhi1[j] * scale);
        }
    }

#if USE_TDM
    const unsigned int kldsBase = (unsigned int)(uintptr_t)&Klds[0][0];
    const unsigned int vldsBase = (unsigned int)(uintptr_t)&Vlds[0][0];
#endif

    // ---- tile prefetch: TDM double-buffered (wave 0 issues), or manual fallback ----
    auto prefetchK = [&](int kb, int buf) {
#if USE_TDM
        if (wave == 0)
            tdm_load_2d(kldsBase + (unsigned int)buf * sizeof(Klds[0]),
                        Kbh + (size_t)kb * DK_,
                        /*tile_d0=*/64, /*tile_d1=*/32, /*stride0=*/64,
                        /*pad_interval: 32 DWORDs=*/4, /*pad_amount: 4 DWORDs=*/3);
#else
        const int key = tid >> 3, d0 = (tid & 7) << 3;
        *(v8bf*)&Klds[buf][key * KSTRIDE + d0] =
            *(const v8bf*)(Kbh + (size_t)(kb + key) * DK_ + d0);
#endif
    };
    auto prefetchV = [&](int kb, int buf) {
#if USE_TDM
        if (wave == 0)
            tdm_load_2d(vldsBase + (unsigned int)buf * sizeof(Vlds[0]),
                        Vth + kb,
                        /*tile_d0=*/32, /*tile_d1=*/64, /*stride0=*/S_LEN,
                        /*pad_interval: 16 DWORDs=*/3, /*pad_amount: 4 DWORDs=*/3);
#else
        const int d = tid >> 2, k0 = (tid & 3) << 3;
        *(v8bf*)&Vlds[buf][d * VSTRIDE + k0] =
            *(const v8bf*)(Vth + (size_t)d * S_LEN + kb + k0);
#endif
    };

    auto loadKB = [&](int buf, int ktile, int chunk) -> v16bf {
        union { v16bf v; v8bf h[2]; } u;
        const __bf16* p = &Klds[buf][(ktile * 16 + l16) * KSTRIDE + chunk * 32 + half * 8];
        u.h[0] = *(const v8bf*)p;
        u.h[1] = *(const v8bf*)(p + 16);
        return u.v;
    };
    auto loadVB = [&](int buf, int dt) -> v16bf {
        union { v16bf v; v8bf h[2]; } u;
        const __bf16* p = &Vlds[buf][(dt * 16 + l16) * VSTRIDE + half * 8];
        u.h[0] = *(const v8bf*)p;
        u.h[1] = *(const v8bf*)(p + 16);
        return u.v;
    };

    // ================= PASS 1: online row max / sum (base-2 domain) =================
    float m[8], l[8];
    #pragma unroll
    for (int r = 0; r < 8; ++r) { m[r] = -1e30f; l[r] = 0.0f; }

    prefetchK(0, 0);
    for (int it = 0; it < NKT; ++it) {
        const int buf = it & 1;
#if USE_TDM
        if (wave == 0) wait_tensor0();
#endif
        __syncthreads();
        if (it + 1 < NKT) prefetchK((it + 1) * 32, buf ^ 1);

        v8f s0 = {}, s1 = {};
        s0 = wmma_bf16(qa0, loadKB(buf, 0, 0), s0);
        s0 = wmma_bf16(qa1, loadKB(buf, 0, 1), s0);
        s1 = wmma_bf16(qa0, loadKB(buf, 1, 0), s1);
        s1 = wmma_bf16(qa1, loadKB(buf, 1, 1), s1);

        #pragma unroll
        for (int r = 0; r < 8; ++r) {
            const float tmax = rowmax16(fmaxf(s0[r], s1[r]));
            const float mn   = fmaxf(m[r], tmax);
            const float corr = fast_exp2(m[r] - mn);
            const float e    = rowsum16(fast_exp2(s0[r] - mn) + fast_exp2(s1[r] - mn));
            l[r] = l[r] * corr + e;
            m[r] = mn;
        }
    }
    float rinv[8];
    #pragma unroll
    for (int r = 0; r < 8; ++r) rinv[r] = fast_rcp(l[r]);

    // ================= PASS 2: normalized probs -> attn, P*V -> context =================
    v8f acc0 = {}, acc1 = {}, acc2 = {}, acc3 = {};
    __bf16* myP = &Plds[wave * 16 * PSTRIDE];

    prefetchK(0, 0);
    prefetchV(0, 0);
    for (int it = 0; it < NKT; ++it) {
        const int buf = it & 1;
        const int kb  = it * 32;
#if USE_TDM
        if (wave == 0) wait_tensor0();
#endif
        __syncthreads();
        if (it + 1 < NKT) { prefetchK((it + 1) * 32, buf ^ 1); prefetchV((it + 1) * 32, buf ^ 1); }

        v8f s0 = {}, s1 = {};
        s0 = wmma_bf16(qa0, loadKB(buf, 0, 0), s0);
        s0 = wmma_bf16(qa1, loadKB(buf, 0, 1), s0);
        s1 = wmma_bf16(qa0, loadKB(buf, 1, 0), s1);
        s1 = wmma_bf16(qa1, loadKB(buf, 1, 1), s1);

        #pragma unroll
        for (int r = 0; r < 8; ++r) {
            const float p0 = fast_exp2(s0[r] - m[r]) * rinv[r];
            const float p1 = fast_exp2(s1[r] - m[r]) * rinv[r];
            const int row  = r + (half << 3);
            float* arow = Ap + (size_t)(qrow0 + row) * S_LEN + kb;
            __builtin_nontemporal_store(p0, arow + l16);        // streaming: keep L2 for K/V
            __builtin_nontemporal_store(p1, arow + 16 + l16);
            myP[row * PSTRIDE + l16]      = (__bf16)p0;
            myP[row * PSTRIDE + 16 + l16] = (__bf16)p1;
        }
        asm volatile("s_wait_dscnt 0" ::: "memory");

        union { v16bf v; v8bf h[2]; } pa;
        const __bf16* pp = &myP[l16 * PSTRIDE + (half << 3)];
        pa.h[0] = *(const v8bf*)pp;
        pa.h[1] = *(const v8bf*)(pp + 16);

        acc0 = wmma_bf16(pa.v, loadVB(buf, 0), acc0);
        acc1 = wmma_bf16(pa.v, loadVB(buf, 1), acc1);
        acc2 = wmma_bf16(pa.v, loadVB(buf, 2), acc2);
        acc3 = wmma_bf16(pa.v, loadVB(buf, 3), acc3);
    }

    #pragma unroll
    for (int r = 0; r < 8; ++r) {
        const int row = qrow0 + r + (half << 3);
        float* cr = Cp + (size_t)row * DK_;
        __builtin_nontemporal_store(acc0[r], cr +  0 + l16);
        __builtin_nontemporal_store(acc1[r], cr + 16 + l16);
        __builtin_nontemporal_store(acc2[r], cr + 32 + l16);
        __builtin_nontemporal_store(acc3[r], cr + 48 + l16);
    }
}

extern "C" void kernel_launch(void* const* d_in, const int* in_sizes, int n_in,
                              void* d_out, int out_size, void* d_ws, size_t ws_size,
                              hipStream_t stream) {
    const float* Q = (const float*)d_in[0];
    const float* K = (const float*)d_in[1];
    const float* V = (const float*)d_in[2];
    // d_in[3] is d_k (==64), compile-time constant.

    float* ctx  = (float*)d_out;                       // [B,H,S,DK]
    float* attn = ctx + (size_t)NBH * S_LEN * DK_;     // [B,H,S,S]

    __bf16* Kbf = (__bf16*)d_ws;                       // [BH,S,DK] bf16
    __bf16* Vt  = Kbf + (size_t)NBH * S_LEN * DK_;     // [BH,DK,S] bf16 (transposed)

    sdpa_prep_kernel<<<dim3(NBH * 32), dim3(256), 0, stream>>>(K, V, Kbf, Vt);
    sdpa_fused_kernel<<<dim3(NBH * (S_LEN / QBLK)), dim3(256), 0, stream>>>(Q, Kbf, Vt, ctx, attn);
}
